// MultiHeadedAttentionv2_62027917689487
// MI455X (gfx1250) — matrix operand order repaired and checked by reading, not compiled, not run on hardware
//
#include <hip/hip_runtime.h>
#include <hip/hip_bf16.h>
#include <math.h>

typedef _Float16 half_t;
typedef __attribute__((ext_vector_type(16))) _Float16 v16h;
typedef __attribute__((ext_vector_type(8)))  float    v8f;
typedef __attribute__((ext_vector_type(4)))  int      v4i;

#define B_   4
#define C_   256
#define H_   128
#define HW_  16384
#define SEG  1048576ull   // N_tok * D per (scale,batch) -- constant across scales

#if defined(__has_builtin)
#if __has_builtin(__builtin_amdgcn_global_load_async_to_lds_b128)
#define HAVE_ASYNC_COPY 1
#endif
#endif

#ifdef HAVE_ASYNC_COPY
typedef __attribute__((address_space(1))) v4i* gptr_v4i;   // global v4i32*
typedef __attribute__((address_space(3))) v4i* lptr_v4i;   // LDS    v4i32*
#endif

// ---------------------------------------------------------------------------
// WMMA fragment helpers (wave32, per CDNA5 ISA 7.12.2 layouts)
// ---------------------------------------------------------------------------
__device__ __forceinline__ int a_kbase(int v, int h) {
  // 16-bit A 16x32: lanes 0-15 hold K 0..7 (v0..3) and 16..23 (v4..7);
  // lanes 16-31 hold K 8..15 and 24..31.
  return (v < 4 ? 2 * v : 16 + 2 * (v - 4)) + h * 8;
}
__device__ __forceinline__ int b_kbase(int v, int h) {
  // 16-bit B 32x16: lanes 0-15 hold K 0..7 and 16..23; lanes 16-31 hold K 8..15 and 24..31.
  return (v < 4 ? 2 * v : 16 + 2 * (v - 4)) + h * 8;
}

// A fragment: element (m,k) at src[m*ms + k*ks]
__device__ __forceinline__ v16h load_a_h16(const half_t* __restrict__ src, long ms, long ks) {
  int l = threadIdx.x & 31, h = l >> 4, m = l & 15;
  v16h a;
#pragma unroll
  for (int v = 0; v < 8; ++v) {
    int kb = a_kbase(v, h);
    a[2 * v]     = src[(long)m * ms + (long)kb * ks];
    a[2 * v + 1] = src[(long)m * ms + (long)(kb + 1) * ks];
  }
  return a;
}
__device__ __forceinline__ v16h load_a_f32(const float* __restrict__ src, long ms, long ks) {
  int l = threadIdx.x & 31, h = l >> 4, m = l & 15;
  v16h a;
#pragma unroll
  for (int v = 0; v < 8; ++v) {
    int kb = a_kbase(v, h);
    a[2 * v]     = (half_t)src[(long)m * ms + (long)kb * ks];
    a[2 * v + 1] = (half_t)src[(long)m * ms + (long)(kb + 1) * ks];
  }
  return a;
}
// B fragment: element (k,n) at src[k*ks + n*ns]
__device__ __forceinline__ v16h load_b_h16(const half_t* __restrict__ src, long ks, long ns) {
  int l = threadIdx.x & 31, h = l >> 4, n = l & 15;
  v16h b;
#pragma unroll
  for (int v = 0; v < 8; ++v) {
    int kb = b_kbase(v, h);
    b[2 * v]     = src[(long)kb * ks + (long)n * ns];
    b[2 * v + 1] = src[(long)(kb + 1) * ks + (long)n * ns];
  }
  return b;
}
__device__ __forceinline__ v16h load_b_f32(const float* __restrict__ src, long ks, long ns) {
  int l = threadIdx.x & 31, h = l >> 4, n = l & 15;
  v16h b;
#pragma unroll
  for (int v = 0; v < 8; ++v) {
    int kb = b_kbase(v, h);
    b[2 * v]     = (half_t)src[(long)kb * ks + (long)n * ns];
    b[2 * v + 1] = (half_t)src[(long)(kb + 1) * ks + (long)n * ns];
  }
  return b;
}

__device__ __forceinline__ v8f wmma_f16(v16h a, v16h b, v8f c) {
  return __builtin_amdgcn_wmma_f32_16x16x32_f16(false, a, false, b, (short)0, c, false, false);
}

// ---------------------------------------------------------------------------
// 1) Fused Q/K/V 1x1-conv GEMM -> f16 token-major per-scale layout
//    16x64 per wave: A (weights) reused across 4 pixel B-tiles.
//    grid: 3 tensors * 4 batch * 16 Mtiles * 256 Ngroups = 49152 waves
// ---------------------------------------------------------------------------
__global__ __launch_bounds__(256) void qkv_proj_kernel(
    const float* __restrict__ x, const float* __restrict__ y,
    const float* __restrict__ Wq, const float* __restrict__ bq,
    const float* __restrict__ Wk, const float* __restrict__ bk,
    const float* __restrict__ Wv, const float* __restrict__ bv,
    half_t* __restrict__ qb, half_t* __restrict__ kbuf, half_t* __restrict__ vbuf) {
  int t = blockIdx.x * 8 + (threadIdx.x >> 5);
  int ntg = t & 255;
  int r   = t >> 8;
  int mt  = r & 15;
  int b   = (r >> 4) & 3;
  int which = r >> 6;

  const float* in   = (which == 0) ? x  : y;
  const float* W    = (which == 0) ? Wq : (which == 1 ? Wk : Wv);
  const float* bias = (which == 0) ? bq : (which == 1 ? bk : bv);
  half_t* outbuf    = (which == 0) ? qb : (which == 1 ? kbuf : vbuf);

  int m0 = mt * 16;      // output channel tile
  int n0 = ntg * 64;     // first pixel of 4 consecutive tiles
  const float* inb = in + (size_t)b * C_ * HW_;

  v8f a0 = {}, a1 = {}, a2 = {}, a3 = {};
  for (int k0 = 0; k0 < C_; k0 += 32) {
    __builtin_prefetch(inb + (size_t)(k0 + 32) * HW_ + n0, 0, 3);
    v16h a  = load_a_f32(W + (size_t)m0 * C_ + k0, C_, 1);
    v16h b0 = load_b_f32(inb + (size_t)k0 * HW_ + n0,      HW_, 1);
    v16h b1 = load_b_f32(inb + (size_t)k0 * HW_ + n0 + 16, HW_, 1);
    v16h b2 = load_b_f32(inb + (size_t)k0 * HW_ + n0 + 32, HW_, 1);
    v16h b3 = load_b_f32(inb + (size_t)k0 * HW_ + n0 + 48, HW_, 1);
    a0 = wmma_f16(a, b0, a0);
    a1 = wmma_f16(a, b1, a1);
    a2 = wmma_f16(a, b2, a2);
    a3 = wmma_f16(a, b3, a3);
  }

  int l = threadIdx.x & 31, hfl = l >> 4, n = l & 15;
  float accs[4][8];
#pragma unroll
  for (int v = 0; v < 8; ++v) { accs[0][v]=a0[v]; accs[1][v]=a1[v]; accs[2][v]=a2[v]; accs[3][v]=a3[v]; }
#pragma unroll
  for (int j = 0; j < 4; ++j) {
    int pix  = n0 + j * 16 + n;
    int hpix = pix >> 7, wpix = pix & 127;
#pragma unroll
    for (int v = 0; v < 8; ++v) {
      int o = m0 + v + 8 * hfl;               // output channel
      float val = accs[j][v] + bias[o];
      int s  = o >> 6, cl = o & 63;           // scale, channel-in-scale
      int lh = s + 1;                          // log2(patch)
      int oy = hpix >> lh, py = hpix & ((1 << lh) - 1);
      int ox = wpix >> lh, px = wpix & ((1 << lh) - 1);
      int outw = H_ >> lh;
      int ntok = oy * outw + ox;
      long D = 64l << (2 * lh);
      long d = (((long)(cl << lh) + py) << lh) + px;
      outbuf[((size_t)s * B_ + b) * SEG + (size_t)ntok * D + d] = (half_t)val;
    }
  }
}

// ---------------------------------------------------------------------------
// 2a) S = scale * Q K^T for one (batch, scale); 16x64 per wave.
// ---------------------------------------------------------------------------
__global__ __launch_bounds__(256) void scores_kernel(
    const half_t* __restrict__ qb, const half_t* __restrict__ kbuf,
    float* __restrict__ S, int b, int s, int N, long D) {
  int nt16 = N >> 4;
  int ng   = nt16 >> 2;                     // groups of 4 column tiles (>=1)
  int total = nt16 * ng;
  int t = blockIdx.x * 8 + (threadIdx.x >> 5);
  if (t >= total) return;                   // whole-wave exit; EXEC stays all-1
  int mt = t / ng, ntg = t % ng;
  const half_t* q = qb   + ((size_t)s * B_ + b) * SEG + (size_t)mt * 16 * D;
  const half_t* k = kbuf + ((size_t)s * B_ + b) * SEG + (size_t)ntg * 64 * D;

  v8f a0 = {}, a1 = {}, a2 = {}, a3 = {};
  for (long d0 = 0; d0 < D; d0 += 32) {
    __builtin_prefetch(q + d0 + 128, 0, 3);
    __builtin_prefetch(k + d0 + 128, 0, 3);
    v16h a  = load_a_h16(q + d0, D, 1);
    v16h b0 = load_b_h16(k + d0,            1, D);
    v16h b1 = load_b_h16(k + 16 * D + d0,   1, D);
    v16h b2 = load_b_h16(k + 32 * D + d0,   1, D);
    v16h b3 = load_b_h16(k + 48 * D + d0,   1, D);
    a0 = wmma_f16(a, b0, a0);
    a1 = wmma_f16(a, b1, a1);
    a2 = wmma_f16(a, b2, a2);
    a3 = wmma_f16(a, b3, a3);
  }
  float rs = rsqrtf((float)D);
  int l = threadIdx.x & 31, hfl = l >> 4, n = l & 15;
  float accs[4][8];
#pragma unroll
  for (int v = 0; v < 8; ++v) { accs[0][v]=a0[v]; accs[1][v]=a1[v]; accs[2][v]=a2[v]; accs[3][v]=a3[v]; }
#pragma unroll
  for (int j = 0; j < 4; ++j) {
#pragma unroll
    for (int v = 0; v < 8; ++v) {
      int m = mt * 16 + v + 8 * hfl;
      S[(size_t)m * N + ntg * 64 + j * 16 + n] = accs[j][v] * rs;
    }
  }
}

// ---------------------------------------------------------------------------
// 2b) row softmax: P(f16) = softmax(S); row staged into LDS via async copy
// ---------------------------------------------------------------------------
__global__ __launch_bounds__(256) void softmax_kernel(
    const float* __restrict__ S, half_t* __restrict__ P, int N) {
  __shared__ float rowbuf[4096];
  __shared__ float red[256];
  int row = blockIdx.x;
  const float* srow = S + (size_t)row * N;

#ifdef HAVE_ASYNC_COPY
  for (int i = threadIdx.x * 4; i < N; i += 1024) {
    __builtin_amdgcn_global_load_async_to_lds_b128(
        (gptr_v4i)(srow + i), (lptr_v4i)(rowbuf + i), 0, 0);
  }
#if __has_builtin(__builtin_amdgcn_s_wait_asynccnt)
  __builtin_amdgcn_s_wait_asynccnt(0);
#else
  asm volatile("s_wait_asynccnt 0x0" ::: "memory");
#endif
#else
  for (int i = threadIdx.x; i < N; i += 256) rowbuf[i] = srow[i];
#endif
  __syncthreads();

  float m = -3.4e38f;
  for (int i = threadIdx.x; i < N; i += 256) m = fmaxf(m, rowbuf[i]);
  red[threadIdx.x] = m; __syncthreads();
  for (int o = 128; o > 0; o >>= 1) {
    if (threadIdx.x < o) red[threadIdx.x] = fmaxf(red[threadIdx.x], red[threadIdx.x + o]);
    __syncthreads();
  }
  m = red[0]; __syncthreads();

  float sum = 0.f;
  for (int i = threadIdx.x; i < N; i += 256) sum += __expf(rowbuf[i] - m);
  red[threadIdx.x] = sum; __syncthreads();
  for (int o = 128; o > 0; o >>= 1) {
    if (threadIdx.x < o) red[threadIdx.x] += red[threadIdx.x + o];
    __syncthreads();
  }
  float inv = 1.0f / red[0];
  for (int i = threadIdx.x; i < N; i += 256)
    P[(size_t)row * N + i] = (half_t)(__expf(rowbuf[i] - m) * inv);
}

// ---------------------------------------------------------------------------
// 2c) O = P V, 16x64 per wave, scatter back to NCHW f16.  1024 waves/launch.
// ---------------------------------------------------------------------------
__global__ __launch_bounds__(256) void pv_kernel(
    const half_t* __restrict__ P, const half_t* __restrict__ vbuf,
    half_t* __restrict__ attn, int b, int s, int N, long D, int lh) {
  int dg = (int)(D >> 6);                  // groups of 4 d-tiles (>=4)
  int t = blockIdx.x * 8 + (threadIdx.x >> 5);
  int mt = t / dg, ntg = t % dg;
  const half_t* v = vbuf + ((size_t)s * B_ + b) * SEG + (size_t)ntg * 64;

  v8f a0 = {}, a1 = {}, a2 = {}, a3 = {};
  for (int k0 = 0; k0 < N; k0 += 32) {
    __builtin_prefetch(v + (size_t)(k0 + 32) * D, 0, 3);
    v16h a  = load_a_h16(P + (size_t)mt * 16 * N + k0, N, 1);
    v16h b0 = load_b_h16(v + (size_t)k0 * D,      D, 1);
    v16h b1 = load_b_h16(v + (size_t)k0 * D + 16, D, 1);
    v16h b2 = load_b_h16(v + (size_t)k0 * D + 32, D, 1);
    v16h b3 = load_b_h16(v + (size_t)k0 * D + 48, D, 1);
    a0 = wmma_f16(a, b0, a0);
    a1 = wmma_f16(a, b1, a1);
    a2 = wmma_f16(a, b2, a2);
    a3 = wmma_f16(a, b3, a3);
  }

  int l = threadIdx.x & 31, hfl = l >> 4, n = l & 15;
  int patch = 1 << lh;
  float accs[4][8];
#pragma unroll
  for (int vv = 0; vv < 8; ++vv) { accs[0][vv]=a0[vv]; accs[1][vv]=a1[vv]; accs[2][vv]=a2[vv]; accs[3][vv]=a3[vv]; }
#pragma unroll
  for (int j = 0; j < 4; ++j) {
#pragma unroll
    for (int vv = 0; vv < 8; ++vv) {
      int tok = mt * 16 + vv + 8 * hfl;
      int d   = ntg * 64 + j * 16 + n;
      int cl  = d >> (2 * lh);
      int rem = d & (patch * patch - 1);
      int py = rem >> lh, px = rem & (patch - 1);
      int outw = H_ >> lh;
      int oy = tok / outw, ox = tok % outw;
      int hh = (oy << lh) + py, wwp = (ox << lh) + px;
      int c  = s * 64 + cl;
      attn[((size_t)b * C_ + c) * HW_ + hh * H_ + wwp] = (half_t)accs[j][vv];
    }
  }
}

// ---------------------------------------------------------------------------
// 3) 3x3 conv as implicit-im2col WMMA GEMM (K = 256*9 = 2304)
//    64x16 per wave: expensive gathered B reused across 4 weight A-tiles.
//    grid: 4 batch * 4 Mgroups * 1024 pixel-tiles = 16384 waves
// ---------------------------------------------------------------------------
__global__ __launch_bounds__(256) void conv3_kernel(
    const half_t* __restrict__ attn, const float* __restrict__ Wo,
    const float* __restrict__ bo, float* __restrict__ z) {
  int t = blockIdx.x * 8 + (threadIdx.x >> 5);
  int nt = t & 1023;
  int r  = t >> 10;
  int mg = r & 3;
  int b  = r >> 2;
  int m0 = mg * 64, n0 = nt * 16;
  int hrow = n0 >> 7;           // 16 contiguous pixels stay in one image row
  int w0   = n0 & 127;
  const half_t* src = attn + (size_t)b * C_ * HW_;
  int l = threadIdx.x & 31, hfl = l >> 4, lm = l & 15;

  v8f c0 = {}, c1 = {}, c2 = {}, c3 = {};
  for (int k0 = 0; k0 < 2304; k0 += 32) {
    __builtin_prefetch(Wo + (size_t)m0 * 2304 + k0 + 64, 0, 3);
    v16h bf;
#pragma unroll
    for (int v = 0; v < 8; ++v) {
#pragma unroll
      for (int e = 0; e < 2; ++e) {
        int k  = k0 + b_kbase(v, hfl) + e;
        int c  = k / 9;
        int rr = k - c * 9;
        int ky = rr / 3, kx = rr - ky * 3;
        int hs = hrow + ky - 1;
        int ws = w0 + lm + kx - 1;
        half_t val = (half_t)0.f;
        if (hs >= 0 && hs < H_ && ws >= 0 && ws < H_)
          val = src[(size_t)c * HW_ + hs * H_ + ws];
        bf[2 * v + e] = val;
      }
    }
    v16h aA = load_a_f32(Wo + (size_t)(m0     ) * 2304 + k0, 2304, 1);
    v16h aB = load_a_f32(Wo + (size_t)(m0 + 16) * 2304 + k0, 2304, 1);
    v16h aC = load_a_f32(Wo + (size_t)(m0 + 32) * 2304 + k0, 2304, 1);
    v16h aD = load_a_f32(Wo + (size_t)(m0 + 48) * 2304 + k0, 2304, 1);
    c0 = wmma_f16(aA, bf, c0);
    c1 = wmma_f16(aB, bf, c1);
    c2 = wmma_f16(aC, bf, c2);
    c3 = wmma_f16(aD, bf, c3);
  }
  float accs[4][8];
#pragma unroll
  for (int v = 0; v < 8; ++v) { accs[0][v]=c0[v]; accs[1][v]=c1[v]; accs[2][v]=c2[v]; accs[3][v]=c3[v]; }
#pragma unroll
  for (int j = 0; j < 4; ++j) {
#pragma unroll
    for (int v = 0; v < 8; ++v) {
      int o = m0 + j * 16 + v + 8 * hfl;
      z[((size_t)b * C_ + o) * HW_ + n0 + lm] = accs[j][v] + bo[o];
    }
  }
}

// ---------------------------------------------------------------------------
// 4) BatchNorm batch-stats (one block per channel) + epilogue
// ---------------------------------------------------------------------------
__global__ __launch_bounds__(256) void bnstats_kernel(
    const float* __restrict__ z, float* __restrict__ stats) {
  __shared__ float s1[256], s2[256];
  int c = blockIdx.x;
  float sum = 0.f, sq = 0.f;
  for (int i = threadIdx.x; i < B_ * HW_; i += 256) {
    int b = i >> 14, p = i & (HW_ - 1);
    float v = z[((size_t)b * C_ + c) * HW_ + p];
    sum += v; sq += v * v;
  }
  s1[threadIdx.x] = sum; s2[threadIdx.x] = sq; __syncthreads();
  for (int o = 128; o > 0; o >>= 1) {
    if (threadIdx.x < o) { s1[threadIdx.x] += s1[threadIdx.x + o]; s2[threadIdx.x] += s2[threadIdx.x + o]; }
    __syncthreads();
  }
  if (threadIdx.x == 0) {
    float mean = s1[0] / (float)(B_ * HW_);
    float var  = s2[0] / (float)(B_ * HW_) - mean * mean;
    stats[c]      = mean;
    stats[C_ + c] = rsqrtf(var + 1e-5f);
  }
}

__global__ __launch_bounds__(256) void finalize_kernel(
    const float* __restrict__ z, const float* __restrict__ stats,
    const float* __restrict__ gamma, const float* __restrict__ beta,
    float* __restrict__ out) {
  size_t i = (size_t)blockIdx.x * 256 + threadIdx.x;   // 16,777,216 total
  int c = (int)((i >> 14) & (C_ - 1));
  float zn = (z[i] - stats[c]) * stats[C_ + c] * gamma[c] + beta[c];
  out[i] = zn >= 0.f ? zn : 0.2f * zn;
}

// ---------------------------------------------------------------------------
// Host launcher
// ---------------------------------------------------------------------------
extern "C" void kernel_launch(void* const* d_in, const int* in_sizes, int n_in,
                              void* d_out, int out_size, void* d_ws, size_t ws_size,
                              hipStream_t stream) {
  const float* x     = (const float*)d_in[0];
  const float* y     = (const float*)d_in[1];
  const float* Wq    = (const float*)d_in[2];
  const float* bq    = (const float*)d_in[3];
  const float* Wk    = (const float*)d_in[4];
  const float* bk    = (const float*)d_in[5];
  const float* Wv    = (const float*)d_in[6];
  const float* bv    = (const float*)d_in[7];
  const float* Wo    = (const float*)d_in[8];
  const float* bo    = (const float*)d_in[9];
  const float* gamma = (const float*)d_in[10];
  const float* beta  = (const float*)d_in[11];
  float* out = (float*)d_out;

  char* ws = (char*)d_ws;
  half_t* qb   = (half_t*)(ws);                    //  33.5 MB  f16 Q tokens
  half_t* kb   = (half_t*)(ws +  33554432);        //  33.5 MB  f16 K tokens
  half_t* vb   = (half_t*)(ws +  67108864);        //  33.5 MB  f16 V tokens
  half_t* attn = (half_t*)(ws + 100663296);        //  33.5 MB  f16 attn NCHW
  float*  S    = (float*) (ws + 134217728);        //  64   MB  scores (fits L2)
  half_t* P    = (half_t*)(ws + 201326592);        //  32   MB  f16 probs
  float*  z    = (float*) (ws + 234881024);        //  64   MB  conv out
  float*  st   = (float*) (ws + 301989888);        //   2   KB  mean/rstd

  qkv_proj_kernel<<<6144, 256, 0, stream>>>(x, y, Wq, bq, Wk, bk, Wv, bv, qb, kb, vb);

  for (int b = 0; b < 4; ++b) {
    for (int s = 0; s < 4; ++s) {
      int  lh = s + 1;
      int  N  = 16384 >> (2 * lh);   // 4096,1024,256,64
      long D  = 64l << (2 * lh);     //  256,1024,4096,16384
      int nt16  = N >> 4;
      int waves = nt16 * (nt16 >> 2);
      scores_kernel<<<(waves + 7) / 8, 256, 0, stream>>>(qb, kb, S, b, s, N, D);
      softmax_kernel<<<N, 256, 0, stream>>>(S, P, N);
      pv_kernel<<<128, 256, 0, stream>>>(P, vb, attn, b, s, N, D, lh);
    }
  }

  conv3_kernel<<<2048, 256, 0, stream>>>(attn, Wo, bo, z);
  bnstats_kernel<<<256, 256, 0, stream>>>(z, st);
  finalize_kernel<<<65536, 256, 0, stream>>>(z, st, gamma, beta, out);
}
